// FNOTransport_52536039964930
// MI455X (gfx1250) — matrix operand order repaired and checked by reading, not compiled, not run on hardware
//
#include <hip/hip_runtime.h>
#include <hip/hip_bf16.h>
#include <math.h>

// ---------------- problem constants (fixed by the reference) ----------------
#define B_      4
#define NX_     4096
#define NW_     64
#define C_      32
#define H_      64
#define W_      64
#define M_      12
#define NFX_    16
#define NFW_    8
#define NFIELD_ 43
#define HID_    128

#define TWO_PI 6.283185307179586f

// output layout (floats): I [B*NX*NW] | phi [B*NX] | J [B*NX*2]
#define PHI_OFF 1048576
#define J_OFF   1064960

// workspace layout (float offsets from d_ws)
#define OFF_ZA   0          // 524288 f32  z ping
#define OFF_ZB   524288     // 524288 f32  z pong
#define OFF_FW   1048576    // 196608 f32  stage-A DFT (complex)
#define OFF_FT   1245184    // 73728  f32  truncated spectrum (complex)
#define OFF_OFT  1318912    // 73728  f32  after spectral mix (complex)
#define OFF_G1   1392640    // 196608 f32  inverse stage-1 (complex)
#define OFF_OMB  1589248    // 32768  f32  ombias [B][NW][128]
#define OFF_END_F32 1622016
// after that: halves — swzW0 (4096), swzW1 (16384), latW0 (2097152)

typedef __attribute__((ext_vector_type(16))) _Float16     v16h;
typedef __attribute__((ext_vector_type(8)))  float        v8f;
typedef __attribute__((ext_vector_type(4)))  unsigned int v4u;
typedef __attribute__((ext_vector_type(8)))  int          v8i_t;
typedef __attribute__((ext_vector_type(4)))  int          v4i_t;

__device__ __forceinline__ float gelu_f(float x){
  return 0.5f * x * (1.0f + erff(x * 0.7071067811865476f));
}
__device__ __forceinline__ float softplus_f(float x){
  return (x > 20.0f) ? x : log1pf(__expf(x));
}

// ------------------------------------------------------------------ TDM helper
// Issue a Tensor Data Mover DMA: nelem f16 elements, global -> LDS.
// D# per CDNA5 ISA: group0 = {count|flags, lds_addr,
// global_addr[31:0], global_addr[56:32] | type=2}; group1 = {data_size=1(2B),
// 1-D tensor: tensor_dim0 = tile_dim0 = nelem, tensor_dim1 = 1, stride = nelem};
// groups 2/3 zero (<=2D), workgroup_mask = 0 (not in a cluster).
__device__ __forceinline__ void tdm_load_f16(const _Float16* gsrc, void* ldst,
                                             unsigned nelem){
  unsigned long long ga = (unsigned long long)(uintptr_t)gsrc;
  unsigned la = (unsigned)(uintptr_t)ldst;     // low 32 bits = LDS byte offset
  v4u g0 = { 1u,                               // count=1, is_restore=0
             la,
             (unsigned)(ga & 0xFFFFFFFFull),
             (unsigned)((ga >> 32) & 0x1FFFFFFull) | (2u << 30) };  // type=2
  v8i_t g1;
  g1[0] = (int)(1u << 16);                               // data_size=1 -> 2 bytes
  g1[1] = (int)((nelem & 0xFFFFu) << 16);                // tensor_dim0[15:0]
  g1[2] = (int)(((nelem >> 16) & 0xFFFFu) | (1u << 16)); // dim0[31:16], tensor_dim1=1
  g1[3] = (int)((nelem & 0xFFFFu) << 16);                // tile_dim0 = nelem
  g1[4] = 0;                                             // tile_dim1/2 unused
  g1[5] = (int)nelem;                                    // tensor_dim0_stride[31:0]
  g1[6] = 0;
  g1[7] = 0;
  v4i_t z4 = {0, 0, 0, 0};
#if defined(__clang_major__) && (__clang_major__ >= 23)
  v8i_t z8 = {0, 0, 0, 0, 0, 0, 0, 0};
  __builtin_amdgcn_tensor_load_to_lds(g0, g1, z4, z4, z8, 0);
#else
  __builtin_amdgcn_tensor_load_to_lds(g0, g1, z4, z4, 0);
#endif
}

// ------------------------------------------------------------------ zero phi/J
__global__ void k_zero(float* out){
  int i = blockIdx.x * 256 + threadIdx.x;          // 49152 total
  out[PHI_OFF + i] = 0.0f;
}

// ------------------------------------------------------------------ encode
__global__ void k_encode(const float* __restrict__ sigma_a, const float* __restrict__ sigma_s,
                         const float* __restrict__ q, const float* __restrict__ bc,
                         const float* __restrict__ params, const float* __restrict__ xx,
                         const float* __restrict__ x_freqs, const float* __restrict__ param_W,
                         const float* __restrict__ param_b, const float* __restrict__ lift_W,
                         const float* __restrict__ lift_b, float* __restrict__ z)
{
  int gid = blockIdx.x * 128 + threadIdx.x;        // B*NX = 16384
  if (gid >= B_ * NX_) return;
  int b = gid / NX_, p = gid % NX_;
  float feats[NFIELD_];
  feats[0] = sigma_a[gid];
  feats[1] = sigma_s[gid];
  feats[2] = q[gid];
#pragma unroll
  for (int j = 0; j < 8; ++j) feats[3 + j] = bc[b * 8 + j];
  float x0 = xx[gid * 2 + 0], x1 = xx[gid * 2 + 1];
#pragma unroll
  for (int f = 0; f < NFX_; ++f){
    float pr = TWO_PI * (x0 * x_freqs[f] + x1 * x_freqs[NFX_ + f]);
    float s, c; __sincosf(pr, &s, &c);
    feats[11 + f] = s;
    feats[11 + NFX_ + f] = c;
  }
  float p0 = params[b * 2 + 0], p1 = params[b * 2 + 1];
  for (int c = 0; c < C_; ++c){
    float acc = lift_b[c] + param_b[c] + p0 * param_W[c] + p1 * param_W[C_ + c];
#pragma unroll
    for (int f = 0; f < NFIELD_; ++f) acc += feats[f] * lift_W[f * C_ + c];
    z[(b * C_ + c) * NX_ + p] = acc;               // [B][C][H*W]
  }
}

// -------------------------------------------------- DFT over w (last axis)
__global__ void k_dftw(const float* __restrict__ z, float* __restrict__ Fw){
  int gid = blockIdx.x * 256 + threadIdx.x;        // B*C*H*M = 98304
  int ky = gid % M_; int t = gid / M_;
  int h = t % H_; int bc = t / H_;
  const float* row = z + (bc * H_ + h) * W_;
  float re = 0.f, im = 0.f;
  for (int w = 0; w < W_; ++w){
    int ph = (ky * w) & 63;
    float s, c; __sincosf(-(TWO_PI / 64.f) * (float)ph, &s, &c);
    float v = row[w];
    re += v * c; im += v * s;
  }
  Fw[gid * 2 + 0] = re; Fw[gid * 2 + 1] = im;
}

// -------------------------------------------------- DFT over h (24 kept rows)
__global__ void k_dfth(const float* __restrict__ Fw, float* __restrict__ FT){
  int gid = blockIdx.x * 256 + threadIdx.x;        // B*C*24*M = 36864
  int ky = gid % M_; int t = gid / M_;
  int kxi = t % 24; int bc = t / 24;
  int kx = (kxi < M_) ? kxi : (kxi + 40);          // 0..11 and 52..63
  float re = 0.f, im = 0.f;
  for (int h = 0; h < H_; ++h){
    float fr = Fw[((bc * H_ + h) * M_ + ky) * 2 + 0];
    float fi = Fw[((bc * H_ + h) * M_ + ky) * 2 + 1];
    int ph = (kx * h) & 63;
    float s, c; __sincosf(-(TWO_PI / 64.f) * (float)ph, &s, &c);
    re += fr * c - fi * s;
    im += fr * s + fi * c;
  }
  FT[gid * 2 + 0] = re; FT[gid * 2 + 1] = im;
}

// -------------------------------------------------- spectral channel mix
__global__ void k_spec(const float* __restrict__ FT, const float* __restrict__ wr,
                       const float* __restrict__ wi, float* __restrict__ OFT, int blk)
{
  int gid = blockIdx.x * 256 + threadIdx.x;        // B*C*24*M = 36864
  int ky = gid % M_; int t = gid / M_;
  int kxi = t % 24; t /= 24;
  int o = t % C_; int b = t / C_;
  int half = (kxi < M_) ? 0 : 1;
  int mx = (kxi < M_) ? kxi : (kxi - M_);
  float re = 0.f, im = 0.f;
  for (int i = 0; i < C_; ++i){
    float fr = FT[(((b * C_ + i) * 24 + kxi) * M_ + ky) * 2 + 0];
    float fi = FT[(((b * C_ + i) * 24 + kxi) * M_ + ky) * 2 + 1];
    int widx = ((((blk * 2 + half) * C_ + i) * C_ + o) * M_ + mx) * M_ + ky;
    float wrv = wr[widx], wiv = wi[widx];
    re += fr * wrv - fi * wiv;
    im += fr * wiv + fi * wrv;
  }
  OFT[(((b * C_ + o) * 24 + kxi) * M_ + ky) * 2 + 0] = re;
  OFT[(((b * C_ + o) * 24 + kxi) * M_ + ky) * 2 + 1] = im;
}

// -------------------------------------------------- inverse DFT over h
__global__ void k_idfth(const float* __restrict__ OFT, float* __restrict__ G1){
  int gid = blockIdx.x * 256 + threadIdx.x;        // B*C*H*M = 98304
  int ky = gid % M_; int t = gid / M_;
  int h = t % H_; int bo = t / H_;
  float re = 0.f, im = 0.f;
  for (int kxi = 0; kxi < 24; ++kxi){
    int kx = (kxi < M_) ? kxi : (kxi + 40);
    float fr = OFT[((bo * 24 + kxi) * M_ + ky) * 2 + 0];
    float fi = OFT[((bo * 24 + kxi) * M_ + ky) * 2 + 1];
    int ph = (kx * h) & 63;
    float s, c; __sincosf((TWO_PI / 64.f) * (float)ph, &s, &c);
    re += fr * c - fi * s;
    im += fr * s + fi * c;
  }
  G1[gid * 2 + 0] = re; G1[gid * 2 + 1] = im;
}

// -------------------------------- inverse DFT over w + pointwise conv + gelu
__global__ void k_idftw_pw(const float* __restrict__ G1, const float* __restrict__ zin,
                           const float* __restrict__ pw_W, const float* __restrict__ pw_b,
                           float* __restrict__ zout, int blk)
{
  int gid = blockIdx.x * 256 + threadIdx.x;        // B*C*H*W = 524288
  int w = gid % W_; int t = gid / W_;
  int h = t % H_; int bo = t / H_;
  int o = bo % C_; int b = bo / C_;
  float s = 0.f;
#pragma unroll
  for (int ky = 0; ky < M_; ++ky){
    float fr = G1[((bo * H_ + h) * M_ + ky) * 2 + 0];
    float fi = G1[((bo * H_ + h) * M_ + ky) * 2 + 1];
    int ph = (ky * w) & 63;
    float sn, cs; __sincosf((TWO_PI / 64.f) * (float)ph, &sn, &cs);
    s += ((ky == 0) ? 1.0f : 2.0f) * (fr * cs - fi * sn);
  }
  s *= (1.0f / (float)(H_ * W_));
  float pw = pw_b[blk * C_ + o];
  for (int i = 0; i < C_; ++i)
    pw += zin[((b * C_ + i) * H_ + h) * W_ + w] * pw_W[(blk * C_ + i) * C_ + o];
  zout[gid] = gelu_f(s + pw);
}

// -------------------------------------------------- omega-feature bias: of@W0[32:]+b0
__global__ void k_ombias(const float* __restrict__ omega, const float* __restrict__ hW0,
                         const float* __restrict__ hb0, float* __restrict__ omb)
{
  int gid = blockIdx.x * 256 + threadIdx.x;        // B*NW*HID = 32768
  int n = gid % HID_; int t = gid / HID_;
  int w = t % NW_; int b = t / NW_;
  float ox = omega[(b * NW_ + w) * 2 + 0], oy = omega[(b * NW_ + w) * 2 + 1];
  float th = atan2f(oy, ox);
  float acc = hb0[n];
#pragma unroll
  for (int f = 0; f < NFW_; ++f){
    float s, c; __sincosf(th * (float)(f + 1), &s, &c);
    acc += s * hW0[(C_ + f) * HID_ + n] + c * hW0[(C_ + NFW_ + f) * HID_ + n];
  }
  omb[(b * NW_ + w) * HID_ + n] = acc;
}

// ------------------------- pre-swizzle W0[:32] and W1 into WMMA B-fragment layout
// B-fragment (16-bit, 32x16): lane L holds N = L&15, K = (L<16?0:8)+e+(e<8?0:8)  for e=0..15
__global__ void k_pack(const float* __restrict__ hW0, const float* __restrict__ hW1,
                       _Float16* __restrict__ swzW0, _Float16* __restrict__ swzW1)
{
  int gid = blockIdx.x * 256 + threadIdx.x;        // 4096 + 16384 = 20480
  if (gid < 4096){
    int e = gid & 15; int lane = (gid >> 4) & 31; int nt = gid >> 9;
    int k = ((lane < 16) ? 0 : 8) + e + ((e < 8) ? 0 : 8);
    int n = nt * 16 + (lane & 15);
    swzW0[gid] = (_Float16)hW0[k * HID_ + n];
  } else {
    int g = gid - 4096;
    int e = g & 15; int lane = (g >> 4) & 31; int ks = (g >> 9) & 3; int nt = g >> 11;
    int k = ks * 32 + ((lane < 16) ? 0 : 8) + e + ((e < 8) ? 0 : 8);
    int n = nt * 16 + (lane & 15);
    swzW1[g] = (_Float16)hW1[k * HID_ + n];
  }
}

// A-fragment (16-bit, 16x32): lane L holds M = L&15;
// element e -> k = klo + (j<4 ? 2j+lo : 16+2(j-4)+lo), klo = (L<16?0:8), j=e>>1, lo=e&1
__device__ __forceinline__ int a_frag_k(int e, int klo){
  int j = e >> 1, lo = e & 1;
  return klo + ((j < 4) ? (2 * j + lo) : (16 + 2 * (j - 4) + lo));
}

// --------------------------- latW0 = latent @ W0[:32,:]  (WMMA, K = C = 32)
__global__ void __launch_bounds__(128) k_latw0(const float* __restrict__ z,
                                               const _Float16* __restrict__ swzW0,
                                               _Float16* __restrict__ latW0)
{
  __shared__ __attribute__((aligned(32))) _Float16 W0s[8 * 32 * 16];  // 8 KB
  if (threadIdx.x < 32){
    tdm_load_f16(swzW0, W0s, 4096);
    __builtin_amdgcn_s_wait_tensorcnt(0);
  }
  __syncthreads();

  int wave = blockIdx.x * 4 + (threadIdx.x >> 5);  // 1024 waves total
  int lane = threadIdx.x & 31;
  int b  = wave >> 8;                              // 256 M-tiles per batch
  int x0 = (wave & 255) * 16;
  int m  = lane & 15;
  int klo = (lane < 16) ? 0 : 8;
  int halfsel = (lane < 16) ? 0 : 8;

  v16h a;
#pragma unroll
  for (int e = 0; e < 16; ++e){
    int k = a_frag_k(e, klo);                      // channel index
    a[e] = (_Float16)z[(b * C_ + k) * NX_ + x0 + m];
  }
#pragma unroll
  for (int nt = 0; nt < 8; ++nt){
    v16h bf = *(const v16h*)(&W0s[(nt * 32 + lane) * 16]);
    v8f c = {0.f,0.f,0.f,0.f,0.f,0.f,0.f,0.f};
    v8f d = __builtin_amdgcn_wmma_f32_16x16x32_f16(false, a, false, bf, (short)0, c, false, false);
    int n = nt * 16 + (lane & 15);
#pragma unroll
    for (int r = 0; r < 8; ++r)
      latW0[(size_t)(b * NX_ + x0 + r + halfsel) * HID_ + n] = (_Float16)d[r];
  }
}

// --------------------------- fused head: gelu(h0) @ W1 -> gelu -> @W2 -> softplus -> moments
__global__ void __launch_bounds__(128) k_head(const _Float16* __restrict__ latW0,
                                              const float* __restrict__ omb,
                                              const _Float16* __restrict__ swzW1g,
                                              const float* __restrict__ hb1,
                                              const float* __restrict__ hW2,
                                              const float* __restrict__ hb2,
                                              const float* __restrict__ omega,
                                              const float* __restrict__ w_omega,
                                              const unsigned char* __restrict__ omask,
                                              float* __restrict__ out)
{
  __shared__ __attribute__((aligned(32))) _Float16 W1s[8 * 4 * 32 * 16]; // 32 KB
  int tid = threadIdx.x;
  if (tid < 32){                                   // wave 0 drives the TDM DMA
    tdm_load_f16(swzW1g, W1s, 16384);
    __builtin_amdgcn_s_wait_tensorcnt(0);
  }
  __syncthreads();

  int b  = blockIdx.x >> 8;                        // 1024 blocks: 4 b x 256 slabs
  int x0 = (blockIdx.x & 255) * 16;
  int waveid = tid >> 5;
  int lane = tid & 31;
  int m = lane & 15;
  int klo = (lane < 16) ? 0 : 8;
  int halfsel = (lane < 16) ? 0 : 8;
  float b2 = hb2[0];

  for (int i = 0; i < 16; ++i){                    // 16 tiles per wave
    int t = waveid * 16 + i;
    int x  = x0 + (t >> 2);
    int w0 = (t & 3) * 16;
    int w  = w0 + m;                               // this lane's row = direction
    const _Float16* latrow = latW0 + (size_t)(b * NX_ + x) * HID_;
    const float*    omrow  = omb   + (size_t)(b * NW_ + w) * HID_;
    if (i + 1 < 16){
      int t2 = waveid * 16 + i + 1;
      __builtin_prefetch(latW0 + (size_t)(b * NX_ + x0 + (t2 >> 2)) * HID_, 0, 0);
    }
    // A fragments: h0 = gelu(latW0[x] + ombias[w]) in fragment layout
    v16h av[4];
#pragma unroll
    for (int ks = 0; ks < 4; ++ks){
#pragma unroll
      for (int e = 0; e < 16; ++e){
        int k = ks * 32 + a_frag_k(e, klo);
        float v = (float)latrow[k] + omrow[k];
        av[ks][e] = (_Float16)gelu_f(v);
      }
    }
    float acc[8] = {0.f,0.f,0.f,0.f,0.f,0.f,0.f,0.f};
#pragma unroll
    for (int nt = 0; nt < 8; ++nt){
      v8f d = {0.f,0.f,0.f,0.f,0.f,0.f,0.f,0.f};
#pragma unroll
      for (int ks = 0; ks < 4; ++ks){
        v16h bf = *(const v16h*)(&W1s[((nt * 4 + ks) * 32 + lane) * 16]);
        d = __builtin_amdgcn_wmma_f32_16x16x32_f16(false, av[ks], false, bf, (short)0, d, false, false);
      }
      int n = nt * 16 + (lane & 15);
      float b1n = hb1[n], w2n = hW2[n];
#pragma unroll
      for (int r = 0; r < 8; ++r)
        acc[r] += gelu_f(d[r] + b1n) * w2n;        // fold 128->1 projection
    }
    // reduce over the 16 lanes (columns) of each half-wave
#pragma unroll
    for (int s = 1; s < 16; s <<= 1){
#pragma unroll
      for (int r = 0; r < 8; ++r)
        acc[r] += __shfl_xor(acc[r], s, 32);
    }
    if ((lane & 15) == 0){                          // lanes 0 (rows 0-7) and 16 (rows 8-15)
      float phi_p = 0.f, jx_p = 0.f, jy_p = 0.f;
#pragma unroll
      for (int r = 0; r < 8; ++r){
        int ww = w0 + r + halfsel;
        float mk = omask[b * NW_ + ww] ? 1.0f : 0.0f;
        float Iv = softplus_f(acc[r] + b2) * mk;
        out[(size_t)(b * NX_ + x) * NW_ + ww] = Iv;
        float wm = w_omega[b * NW_ + ww] * mk;
        phi_p += Iv * wm;
        jx_p  += Iv * wm * omega[(b * NW_ + ww) * 2 + 0];
        jy_p  += Iv * wm * omega[(b * NW_ + ww) * 2 + 1];
      }
      atomicAdd(&out[PHI_OFF + b * NX_ + x], phi_p);
      atomicAdd(&out[J_OFF + (b * NX_ + x) * 2 + 0], jx_p);
      atomicAdd(&out[J_OFF + (b * NX_ + x) * 2 + 1], jy_p);
    }
  }
}

// ---------------------------------------------------------------- launch
extern "C" void kernel_launch(void* const* d_in, const int* in_sizes, int n_in,
                              void* d_out, int out_size, void* d_ws, size_t ws_size,
                              hipStream_t stream)
{
  (void)in_sizes; (void)n_in; (void)out_size; (void)ws_size;
  const float* sigma_a  = (const float*)d_in[0];
  const float* sigma_s  = (const float*)d_in[1];
  const float* q        = (const float*)d_in[2];
  const float* bc       = (const float*)d_in[3];
  const float* params   = (const float*)d_in[4];
  const float* xx       = (const float*)d_in[5];
  const float* omega    = (const float*)d_in[6];
  const float* w_omega  = (const float*)d_in[7];
  const unsigned char* omask = (const unsigned char*)d_in[8];
  // d_in[9], d_in[10] = H, W (compile-time 64)
  const float* x_freqs  = (const float*)d_in[11];
  const float* param_W  = (const float*)d_in[12];
  const float* param_b  = (const float*)d_in[13];
  const float* lift_W   = (const float*)d_in[14];
  const float* lift_b   = (const float*)d_in[15];
  const float* spec_wr  = (const float*)d_in[16];
  const float* spec_wi  = (const float*)d_in[17];
  const float* pw_W     = (const float*)d_in[18];
  const float* pw_b     = (const float*)d_in[19];
  const float* h_W0     = (const float*)d_in[20];
  const float* h_b0     = (const float*)d_in[21];
  const float* h_W1     = (const float*)d_in[22];
  const float* h_b1     = (const float*)d_in[23];
  const float* h_W2     = (const float*)d_in[24];
  const float* h_b2     = (const float*)d_in[25];

  float* ws = (float*)d_ws;
  float* zA  = ws + OFF_ZA;
  float* zB  = ws + OFF_ZB;
  float* Fw  = ws + OFF_FW;
  float* FT  = ws + OFF_FT;
  float* OFT = ws + OFF_OFT;
  float* G1  = ws + OFF_G1;
  float* OMB = ws + OFF_OMB;
  _Float16* halfbase = (_Float16*)(ws + OFF_END_F32);
  _Float16* swzW0 = halfbase;            // 4096 halves
  _Float16* swzW1 = halfbase + 4096;     // 16384 halves
  _Float16* latW0 = halfbase + 20480;    // 2097152 halves
  float* out = (float*)d_out;

  k_zero  <<<192, 256, 0, stream>>>(out);
  k_encode<<<128, 128, 0, stream>>>(sigma_a, sigma_s, q, bc, params, xx,
                                    x_freqs, param_W, param_b, lift_W, lift_b, zA);
  k_pack  <<<80, 256, 0, stream>>>(h_W0, h_W1, swzW0, swzW1);
  k_ombias<<<128, 256, 0, stream>>>(omega, h_W0, h_b0, OMB);

  float* zin = zA; float* zout = zB;
  for (int blk = 0; blk < 4; ++blk){
    k_dftw    <<<384,  256, 0, stream>>>(zin, Fw);
    k_dfth    <<<144,  256, 0, stream>>>(Fw, FT);
    k_spec    <<<144,  256, 0, stream>>>(FT, spec_wr, spec_wi, OFT, blk);
    k_idfth   <<<384,  256, 0, stream>>>(OFT, G1);
    k_idftw_pw<<<2048, 256, 0, stream>>>(G1, zin, pw_W, pw_b, zout, blk);
    float* tmp = zin; zin = zout; zout = tmp;
  }
  // after 4 swaps the latent is back in zA (== zin)
  k_latw0<<<256, 128, 0, stream>>>(zin, swzW0, latW0);
  k_head <<<1024, 128, 0, stream>>>(latW0, OMB, swzW1, h_b1, h_W2, h_b2,
                                    omega, w_omega, omask, out);
}
